// MambaBlock_1726576855425
// MI455X (gfx1250) — 8-bit matrix operand order repaired and checked by reading, compiled, not run on hardware
//
#include <hip/hip_runtime.h>
#include <math.h>

typedef __attribute__((ext_vector_type(8))) int v8i;

#define DIM      1024
#define D_INNER  2048
#define D_CONV   4
#define DT_RANK  64
#define D_STATE  16
#define SEQ      2048
#define NPROJ    (DT_RANK + 2*D_STATE)

__global__ void wstats_kernel(const float* __restrict__ w, int n, float* __restrict__ sums) {
  float s = 0.f, sa = 0.f;
  for (int i = blockIdx.x * blockDim.x + threadIdx.x; i < n; i += gridDim.x * blockDim.x) {
    float v = w[i]; s += v; sa += fabsf(v);
  }
  __shared__ float ls[256], la[256];
  int t = threadIdx.x;
  ls[t] = s; la[t] = sa; __syncthreads();
  for (int o = 128; o > 0; o >>= 1) {
    if (t < o) { ls[t] += ls[t + o]; la[t] += la[t + o]; }
    __syncthreads();
  }
  if (t == 0) { atomicAdd(&sums[0], ls[0]); atomicAdd(&sums[1], la[0]); }
}

__global__ void wquant_kernel(const float* __restrict__ w, int n, const float* __restrict__ sums,
                              signed char* __restrict__ w8, float* __restrict__ wscale) {
  float inv_n = 1.f / (float)n;
  if (blockIdx.x == 0 && threadIdx.x == 0) *wscale = sums[1] * inv_n;
  float mean = sums[0] * inv_n;
  for (int i = blockIdx.x * blockDim.x + threadIdx.x; i < n; i += gridDim.x * blockDim.x) {
    float v = w[i] - mean;
    w8[i] = (v > 0.f) ? (signed char)1 : ((v < 0.f) ? (signed char)-1 : (signed char)0);
  }
}

__global__ void actquant_kernel(const float* __restrict__ x, int K, int row_stride,
                                signed char* __restrict__ xq, float* __restrict__ inv_scale) {
  int m = blockIdx.x;
  const float* row = x + (size_t)m * row_stride;
  float ss = 0.f, mx = 0.f;
  for (int k = threadIdx.x; k < K; k += blockDim.x) {
    float v = row[k]; ss += v * v; mx = fmaxf(mx, fabsf(v));
  }
  __shared__ float s_ss[256], s_mx[256];
  int t = threadIdx.x;
  s_ss[t] = ss; s_mx[t] = mx; __syncthreads();
  for (int o = 128; o > 0; o >>= 1) {
    if (t < o) { s_ss[t] += s_ss[t + o]; s_mx[t] = fmaxf(s_mx[t], s_mx[t + o]); }
    __syncthreads();
  }
  float rms   = rsqrtf(s_ss[0] / (float)K + 1e-12f);
  float scale = 127.f / fmaxf(s_mx[0] * rms, 1e-5f);
  if (t == 0) inv_scale[m] = 1.f / scale;
  signed char* out = xq + (size_t)m * K;
  for (int k = threadIdx.x; k < K; k += blockDim.x) {
    float v = rintf(row[k] * rms * scale);
    v = fminf(fmaxf(v, -128.f), 127.f);
    out[k] = (signed char)v;
  }
}

__device__ __forceinline__ int a_koff(int v, int half) {
  int vv = v & 3;
  return ((v >> 2) << 5) + (half << 4) + ((v & 3) << 2);
}
__device__ __forceinline__ int b_koff(int v, int half) {
  return ((v >> 2) << 5) + (half << 4) + ((v & 3) << 2);
}

template <int MT, int NT>
__global__ void gemm_iu8_kernel(const signed char* __restrict__ A,
                                const signed char* __restrict__ Bw,
                                const float* __restrict__ inv_scale,
                                const float* __restrict__ wscale,
                                float* __restrict__ Cout,
                                int K, int out_stride, int act_softplus) {
  const int lane = threadIdx.x;
  const int half = lane >> 4;
  const int idx  = lane & 15;
  const int tn0 = blockIdx.x * NT;
  const int tm0 = blockIdx.y * MT;

  const signed char* Arow[MT];
  const signed char* Brow[NT];
#pragma unroll
  for (int mi = 0; mi < MT; mi++) Arow[mi] = A  + (size_t)((tm0 + mi) * 16 + idx) * K;
#pragma unroll
  for (int ni = 0; ni < NT; ni++) Brow[ni] = Bw + (size_t)((tn0 + ni) * 16 + idx) * K;

  v8i acc[MT][NT];
#pragma unroll
  for (int mi = 0; mi < MT; mi++)
#pragma unroll
    for (int ni = 0; ni < NT; ni++) acc[mi][ni] = (v8i){0, 0, 0, 0, 0, 0, 0, 0};

  for (int k0 = 0; k0 < K; k0 += 64) {
    v8i a[MT], b[NT];
#pragma unroll
    for (int mi = 0; mi < MT; mi++)
#pragma unroll
      for (int v = 0; v < 8; v++)
        a[mi][v] = *(const int*)(Arow[mi] + k0 + a_koff(v, half));
#pragma unroll
    for (int ni = 0; ni < NT; ni++)
#pragma unroll
      for (int v = 0; v < 8; v++)
        b[ni][v] = *(const int*)(Brow[ni] + k0 + b_koff(v, half));
#pragma unroll
    for (int mi = 0; mi < MT; mi++)
#pragma unroll
      for (int ni = 0; ni < NT; ni++)
        acc[mi][ni] = __builtin_amdgcn_wmma_i32_16x16x64_iu8(
            true, a[mi], true, b[ni], acc[mi][ni], false, false);
  }

  const float ws = *wscale;
#pragma unroll
  for (int mi = 0; mi < MT; mi++)
#pragma unroll
    for (int ni = 0; ni < NT; ni++) {
      const int n = (tn0 + ni) * 16 + idx;
#pragma unroll
      for (int g = 0; g < 8; g++) {
        int m = (tm0 + mi) * 16 + g + half * 8;
        float v = (float)acc[mi][ni][g] * ws * inv_scale[m];
        if (act_softplus) v = (v > 20.f) ? v : log1pf(expf(v));
        Cout[(size_t)m * out_stride + n] = v;
      }
    }
}

__global__ void conv_silu_kernel(const float* __restrict__ xz, const float* __restrict__ conv_w,
                                 const float* __restrict__ conv_b, float* __restrict__ xs) {
  int i = blockIdx.x * blockDim.x + threadIdx.x;
  if (i >= SEQ * D_INNER) return;
  int l = i / D_INNER, e = i % D_INNER;
  float acc = conv_b[e];
#pragma unroll
  for (int j = 0; j < D_CONV; j++) {
    int ls = l - (D_CONV - 1) + j;
    if (ls >= 0) acc += xz[(size_t)ls * (2 * D_INNER) + e] * conv_w[e * D_CONV + j];
  }
  xs[i] = acc / (1.f + expf(-acc));
}

__global__ void scan_kernel(const float* __restrict__ delta, const float* __restrict__ xs,
                            const float* __restrict__ dBC, const float* __restrict__ A_log,
                            const float* __restrict__ Dvec, float* __restrict__ y) {
  int n = threadIdx.x & 15;
  int e = blockIdx.x * (blockDim.x >> 4) + (threadIdx.x >> 4);
  float A  = -expf(A_log[e * D_STATE + n]);
  float Dv = Dvec[e];
  float h  = 0.f;
  for (int l = 0; l < SEQ; l++) {
    float dl = delta[(size_t)l * D_INNER + e];
    float xv = xs[(size_t)l * D_INNER + e];
    float Bv = dBC[l * NPROJ + DT_RANK + n];
    float Cv = dBC[l * NPROJ + DT_RANK + D_STATE + n];
    h = expf(dl * A) * h + dl * Bv * xv;
    float c = h * Cv;
    c += __shfl_xor(c, 1);
    c += __shfl_xor(c, 2);
    c += __shfl_xor(c, 4);
    c += __shfl_xor(c, 8);
    if (n == 0) y[(size_t)l * D_INNER + e] = c + Dv * xv;
  }
}

__global__ void gate_kernel(const float* __restrict__ xz, const float* __restrict__ y,
                            float* __restrict__ u) {
  int i = blockIdx.x * blockDim.x + threadIdx.x;
  if (i >= SEQ * D_INNER) return;
  int l = i / D_INNER, e = i % D_INNER;
  float z = xz[(size_t)l * (2 * D_INNER) + D_INNER + e];
  u[i] = (z / (1.f + expf(-z))) * y[i];
}

extern "C" void kernel_launch(void* const* d_in, const int* in_sizes, int n_in,
                              void* d_out, int out_size, void* d_ws, size_t ws_size,
                              hipStream_t stream) {
  const float* x      = (const float*)d_in[0];
  const float* W_in   = (const float*)d_in[1];
  const float* conv_w = (const float*)d_in[2];
  const float* conv_b = (const float*)d_in[3];
  const float* W_x    = (const float*)d_in[4];
  const float* W_dt   = (const float*)d_in[5];
  const float* A_log  = (const float*)d_in[6];
  const float* Dv     = (const float*)d_in[7];
  const float* W_out  = (const float*)d_in[8];
  float* out = (float*)d_out;

  char* p = (char*)d_ws;
  auto take = [&](size_t bytes) { char* r = p; p += (bytes + 255) & ~(size_t)255; return r; };
  float*       sums   = (float*)      take(8 * sizeof(float));
  float*       wsc    = (float*)      take(4 * sizeof(float));
  float*       inv    = (float*)      take(SEQ * sizeof(float));
  signed char* W_in8  = (signed char*)take((size_t)2 * D_INNER * DIM);
  signed char* W_x8   = (signed char*)take((size_t)NPROJ * D_INNER);
  signed char* W_dt8  = (signed char*)take((size_t)D_INNER * DT_RANK);
  signed char* W_out8 = (signed char*)take((size_t)DIM * D_INNER);
  signed char* actq   = (signed char*)take((size_t)SEQ * D_INNER);
  float*       xz     = (float*)      take((size_t)SEQ * 2 * D_INNER * 4);
  float*       xs     = (float*)      take((size_t)SEQ * D_INNER * 4);
  float*       dBC    = (float*)      take((size_t)SEQ * NPROJ * 4);
  float*       delta  = (float*)      take((size_t)SEQ * D_INNER * 4);
  float*       y      = (float*)      take((size_t)SEQ * D_INNER * 4);
  (void)ws_size; (void)n_in; (void)in_sizes; (void)out_size;

  hipMemsetAsync(sums, 0, 8 * sizeof(float), stream);

  wstats_kernel<<<256, 256, 0, stream>>>(W_in,  2 * D_INNER * DIM, sums + 0);
  wstats_kernel<<<256, 256, 0, stream>>>(W_x,   NPROJ * D_INNER,   sums + 2);
  wstats_kernel<<<256, 256, 0, stream>>>(W_dt,  D_INNER * DT_RANK, sums + 4);
  wstats_kernel<<<256, 256, 0, stream>>>(W_out, DIM * D_INNER,     sums + 6);
  wquant_kernel<<<256, 256, 0, stream>>>(W_in,  2 * D_INNER * DIM, sums + 0, W_in8,  wsc + 0);
  wquant_kernel<<<256, 256, 0, stream>>>(W_x,   NPROJ * D_INNER,   sums + 2, W_x8,   wsc + 1);
  wquant_kernel<<<256, 256, 0, stream>>>(W_dt,  D_INNER * DT_RANK, sums + 4, W_dt8,  wsc + 2);
  wquant_kernel<<<256, 256, 0, stream>>>(W_out, DIM * D_INNER,     sums + 6, W_out8, wsc + 3);

  actquant_kernel<<<SEQ, 256, 0, stream>>>(x, DIM, DIM, actq, inv);
  gemm_iu8_kernel<2, 4><<<dim3(2 * D_INNER / 64, SEQ / 32), 32, 0, stream>>>(
      actq, W_in8, inv, wsc + 0, xz, DIM, 2 * D_INNER, 0);

  conv_silu_kernel<<<(SEQ * D_INNER + 255) / 256, 256, 0, stream>>>(xz, conv_w, conv_b, xs);

  actquant_kernel<<<SEQ, 256, 0, stream>>>(xs, D_INNER, D_INNER, actq, inv);
  gemm_iu8_kernel<2, 3><<<dim3(NPROJ / 48, SEQ / 32), 32, 0, stream>>>(
      actq, W_x8, inv, wsc + 1, dBC, D_INNER, NPROJ, 0);

  actquant_kernel<<<SEQ, 256, 0, stream>>>(dBC, DT_RANK, NPROJ, actq, inv);
  gemm_iu8_kernel<2, 4><<<dim3(D_INNER / 64, SEQ / 32), 32, 0, stream>>>(
      actq, W_dt8, inv, wsc + 2, delta, DT_RANK, D_INNER,  1);

  scan_kernel<<<D_INNER / 16, 256, 0, stream>>>(delta, xs, dBC, A_log, Dv, y);

  gate_kernel<<<(SEQ * D_INNER + 255) / 256, 256, 0, stream>>>(xz, y, xs);
  actquant_kernel<<<SEQ, 256, 0, stream>>>(xs, D_INNER, D_INNER, actq, inv);
  gemm_iu8_kernel<2, 4><<<dim3(DIM / 64, SEQ / 32), 32, 0, stream>>>(
      actq, W_out8, inv, wsc + 3, out, D_INNER, DIM, 0);
}
